// DeformableStripAttention_68461778698537
// MI455X (gfx1250) — compile-verified, matrix operand order checked by reading, not run on hardware
//
#include <hip/hip_runtime.h>
#include <hip/hip_bf16.h>

typedef __bf16 bf16_t;
typedef __attribute__((ext_vector_type(8)))  __bf16 v8bf;
typedef __attribute__((ext_vector_type(16))) __bf16 v16bf;
typedef __attribute__((ext_vector_type(8)))  float  v8f;

constexpr int BB = 2;          // batch
constexpr int CC = 256;        // channels
constexpr int HH = 64, WW = 64;
constexpr int PP = HH * WW;    // 4096 pixels
constexpr int NH = 4;          // heads
constexpr int HDIM = 64;       // dims per head
constexpr int NS = 4;          // strips
constexpr float SCALE = 0.125f;      // HDIM^-0.5
constexpr float GN_EPS = 1e-5f;

// ---------------------------------------------------------------------------
// helpers
// ---------------------------------------------------------------------------
__device__ __forceinline__ float wred_sum(float v) {
#pragma unroll
  for (int o = 16; o > 0; o >>= 1) v += __shfl_xor(v, o, 32);
  return v;
}
__device__ __forceinline__ float wred_max(float v) {
#pragma unroll
  for (int o = 16; o > 0; o >>= 1) v = fmaxf(v, __shfl_xor(v, o, 32));
  return v;
}
__device__ __forceinline__ float4 wred_sum4(float4 v) {
#pragma unroll
  for (int o = 16; o > 0; o >>= 1) {
    v.x += __shfl_xor(v.x, o, 32);
    v.y += __shfl_xor(v.y, o, 32);
    v.z += __shfl_xor(v.z, o, 32);
    v.w += __shfl_xor(v.w, o, 32);
  }
  return v;
}

// Load a CDNA5 16-bit A/B fragment whose per-lane K values are CONTIGUOUS in
// memory (row-major A with k-contiguous rows / pixel-major B with c-contiguous
// columns). Layout (ISA 7.12.2): elements j=0..7 -> k0 + (lane>=16?8:0) + 0..7,
// elements j=8..15 -> k0 + 16 + (lane>=16?8:0) + 0..7.  => two b128 loads.
__device__ __forceinline__ v16bf load_frag_kcontig(const bf16_t* __restrict__ base,
                                                   int k0, int lane) {
  int koff = k0 + ((lane >> 4) << 3);
  v8bf lo = *(const v8bf*)(base + koff);
  v8bf hi = *(const v8bf*)(base + koff + 16);
  return __builtin_shufflevector(lo, hi, 0, 1, 2, 3, 4, 5, 6, 7,
                                 8, 9, 10, 11, 12, 13, 14, 15);
}

// ---------------------------------------------------------------------------
// LDS-tiled transpose + convert: x [b,c,p] f32  ->  xT [b,p,c] bf16
// grid (PP/32, CC/32, BB), block (32,8)
// ---------------------------------------------------------------------------
__global__ void k_transpose_bf16(const float* __restrict__ x,
                                 bf16_t* __restrict__ xT) {
  __shared__ float tile[32][33];
  int p0 = blockIdx.x * 32, c0 = blockIdx.y * 32, b = blockIdx.z;
  int tx = threadIdx.x, ty = threadIdx.y;
#pragma unroll
  for (int i = 0; i < 4; ++i) {
    int c = c0 + ty + i * 8;
    tile[ty + i * 8][tx] = x[((size_t)(b * CC + c)) * PP + p0 + tx];
  }
  __syncthreads();
#pragma unroll
  for (int i = 0; i < 4; ++i) {
    int p = p0 + ty + i * 8;
    xT[((size_t)(b * PP + p)) * CC + c0 + tx] = (bf16_t)tile[tx][ty + i * 8];
  }
}

// ---------------------------------------------------------------------------
// fp32 -> bf16 conversion (weights)
// ---------------------------------------------------------------------------
__global__ void k_f32_to_bf16(const float* __restrict__ src,
                              bf16_t* __restrict__ dst, int n) {
  int i = blockIdx.x * blockDim.x + threadIdx.x;
  int stride = gridDim.x * blockDim.x;
  for (; i < n; i += stride) dst[i] = (bf16_t)src[i];
}

// ---------------------------------------------------------------------------
// per-(b,c) spatial mean
// ---------------------------------------------------------------------------
__global__ void k_avg(const float* __restrict__ x, float* __restrict__ avg) {
  __shared__ float red[256];
  int bc = blockIdx.x;
  const float* src = x + (size_t)bc * PP;
  float s = 0.f;
  for (int i = threadIdx.x; i < PP; i += 256) s += src[i];
  red[threadIdx.x] = s;
  __syncthreads();
  for (int o = 128; o > 0; o >>= 1) {
    if (threadIdx.x < o) red[threadIdx.x] += red[threadIdx.x + o];
    __syncthreads();
  }
  if (threadIdx.x == 0) avg[bc] = red[0] * (1.f / PP);
}

// ---------------------------------------------------------------------------
// dirs = normalize((avg @ dir_W^T + dir_b).reshape(..,2)); 1 block, 64 threads
// ---------------------------------------------------------------------------
__global__ void k_dirs(const float* __restrict__ avg, const float* __restrict__ dirW,
                       const float* __restrict__ dirb, float* __restrict__ dirs) {
  __shared__ float raw[64];
  int i = threadIdx.x;
  int b = i >> 5, r = i & 31;
  float s = dirb[r];
  const float* a = avg + b * CC;
  const float* w = dirW + r * CC;
  for (int c = 0; c < CC; ++c) s += a[c] * w[c];
  raw[i] = s;
  __syncthreads();
  if ((r & 1) == 0) {
    float vx = raw[i], vy = raw[i + 1];
    float n = fmaxf(sqrtf(vx * vx + vy * vy), 1e-6f);
    dirs[i]     = vx / n;
    dirs[i + 1] = vy / n;
  }
}

// ---------------------------------------------------------------------------
// WMMA GEMM, pixel-major output: YT[b,p,o] = sum_c W[o,c] * XT[b,p,c]
// wave computes a 16(o) x 64(p) strip: A fragment reused across 4 WMMAs.
// grid: 256 blocks x 256 threads (2048 strips = BB*16*64)
// ---------------------------------------------------------------------------
__global__ void k_gemm_qkv(const bf16_t* __restrict__ Wbf,   // [CC,CC] (o,c)
                           const bf16_t* __restrict__ XT,    // [BB,PP,CC]
                           float* __restrict__ YT) {         // [BB,PP,CC]
  int wave = threadIdx.x >> 5;
  int lane = threadIdx.x & 31;
  int strip = blockIdx.x * 8 + wave;       // BB*16*64
  int b  = strip >> 10;
  int ot = (strip >> 6) & 15;
  int ps = strip & 63;                     // 64-pixel strip index
  int m = lane & 15;
  const bf16_t* Arow = Wbf + (size_t)(ot * 16 + m) * CC;
  const bf16_t* Xb   = XT + (size_t)b * PP * CC;

  v8f acc0 = {}, acc1 = {}, acc2 = {}, acc3 = {};
  int p0 = ps * 64 + m;                    // per-lane pixel col for tile 0
#pragma unroll
  for (int kc = 0; kc < 8; ++kc) {
    int k0 = kc * 32;
    v16bf af = load_frag_kcontig(Arow, k0, lane);
    v16bf b0 = load_frag_kcontig(Xb + (size_t)(p0)      * CC, k0, lane);
    v16bf b1 = load_frag_kcontig(Xb + (size_t)(p0 + 16) * CC, k0, lane);
    v16bf b2 = load_frag_kcontig(Xb + (size_t)(p0 + 32) * CC, k0, lane);
    v16bf b3 = load_frag_kcontig(Xb + (size_t)(p0 + 48) * CC, k0, lane);
    acc0 = __builtin_amdgcn_wmma_f32_16x16x32_bf16(false, af, false, b0, (short)0, acc0, false, false);
    acc1 = __builtin_amdgcn_wmma_f32_16x16x32_bf16(false, af, false, b1, (short)0, acc1, false, false);
    acc2 = __builtin_amdgcn_wmma_f32_16x16x32_bf16(false, af, false, b2, (short)0, acc2, false, false);
    acc3 = __builtin_amdgcn_wmma_f32_16x16x32_bf16(false, af, false, b3, (short)0, acc3, false, false);
  }
  // D layout: lane n = p, VGPR r -> o = ot*16 + r + (lane>=16 ? 8 : 0)
  int obase = ot * 16 + ((lane >> 4) << 3);
  float* Yb = YT + (size_t)b * PP * CC;
  v8f accs[4] = {acc0, acc1, acc2, acc3};
#pragma unroll
  for (int tt = 0; tt < 4; ++tt) {
    float* yrow = Yb + (size_t)(p0 + tt * 16) * CC + obase;
    v8f a = accs[tt];
    *(float4*)(yrow)     = float4{a[0], a[1], a[2], a[3]};
    *(float4*)(yrow + 4) = float4{a[4], a[5], a[6], a[7]};
  }
}

// ---------------------------------------------------------------------------
// final projection: out[b,o,p] = Wo @ gn + bo + x   (channel-major store)
// ---------------------------------------------------------------------------
__global__ void k_gemm_out(const bf16_t* __restrict__ Wbf, const bf16_t* __restrict__ GT,
                           const float* __restrict__ bo, const float* __restrict__ xres,
                           float* __restrict__ out) {
  int wave = threadIdx.x >> 5;
  int lane = threadIdx.x & 31;
  int strip = blockIdx.x * 8 + wave;
  int b  = strip >> 10;
  int ot = (strip >> 6) & 15;
  int ps = strip & 63;
  int m = lane & 15;
  const bf16_t* Arow = Wbf + (size_t)(ot * 16 + m) * CC;
  const bf16_t* Gb   = GT + (size_t)b * PP * CC;

  v8f acc0 = {}, acc1 = {}, acc2 = {}, acc3 = {};
  int p0 = ps * 64 + m;
#pragma unroll
  for (int kc = 0; kc < 8; ++kc) {
    int k0 = kc * 32;
    v16bf af = load_frag_kcontig(Arow, k0, lane);
    v16bf b0 = load_frag_kcontig(Gb + (size_t)(p0)      * CC, k0, lane);
    v16bf b1 = load_frag_kcontig(Gb + (size_t)(p0 + 16) * CC, k0, lane);
    v16bf b2 = load_frag_kcontig(Gb + (size_t)(p0 + 32) * CC, k0, lane);
    v16bf b3 = load_frag_kcontig(Gb + (size_t)(p0 + 48) * CC, k0, lane);
    acc0 = __builtin_amdgcn_wmma_f32_16x16x32_bf16(false, af, false, b0, (short)0, acc0, false, false);
    acc1 = __builtin_amdgcn_wmma_f32_16x16x32_bf16(false, af, false, b1, (short)0, acc1, false, false);
    acc2 = __builtin_amdgcn_wmma_f32_16x16x32_bf16(false, af, false, b2, (short)0, acc2, false, false);
    acc3 = __builtin_amdgcn_wmma_f32_16x16x32_bf16(false, af, false, b3, (short)0, acc3, false, false);
  }
  int hi = (lane >> 4) << 3;
  v8f accs[4] = {acc0, acc1, acc2, acc3};
#pragma unroll
  for (int tt = 0; tt < 4; ++tt) {
    int pcol = p0 + tt * 16;
#pragma unroll
    for (int r = 0; r < 8; ++r) {
      int oc = ot * 16 + r + hi;
      size_t idx = (size_t)(b * CC + oc) * PP + pcol;
      out[idx] = accs[tt][r] + bo[oc] + xres[idx];
    }
  }
}

// ---------------------------------------------------------------------------
// deformable strip attention on pixel-major Q/K/V: wave per query, lanes=samples
// ---------------------------------------------------------------------------
__global__ void k_attn(const float* __restrict__ QT, const float* __restrict__ KT,
                       const float* __restrict__ VT, const float* __restrict__ dirs,
                       float* __restrict__ OT) {
  int wave = threadIdx.x >> 5;
  int lane = threadIdx.x & 31;
  int q = blockIdx.x * 8 + wave;            // 0 .. BB*NH*PP-1
  int b = q >> 14;
  int h = (q >> 12) & 3;
  int p = q & 4095;
  int iy = p >> 6, ix = p & 63;

  int s = lane >> 3, m = lane & 7;
  float dx = dirs[((b * NH + h) * NS + s) * 2 + 0];
  float dy = dirs[((b * NH + h) * NS + s) * 2 + 1];
  float t = -0.5f + (float)m * (1.0f / 7.0f);
  float bxn = -1.f + (float)ix * (2.f / 63.f);
  float byn = -1.f + (float)iy * (2.f / 63.f);
  float gx = fminf(fmaxf(bxn + dx * t, -1.f), 1.f);
  float gy = fminf(fmaxf(byn + dy * t, -1.f), 1.f);
  float px = (gx + 1.f) * 0.5f * (WW - 1);
  float py = (gy + 1.f) * 0.5f * (HH - 1);
  float x0f = floorf(px), y0f = floorf(py);
  float fx = px - x0f, fy = py - y0f;
  int x0 = min(max((int)x0f, 0), WW - 1);
  int x1 = min(x0 + 1, WW - 1);
  int y0 = min(max((int)y0f, 0), HH - 1);
  int y1 = min(y0 + 1, HH - 1);
  float w00 = (1.f - fx) * (1.f - fy);
  float w01 = fx * (1.f - fy);
  float w10 = (1.f - fx) * fy;
  float w11 = fx * fy;
  int i00 = y0 * WW + x0, i01 = y0 * WW + x1;
  int i10 = y1 * WW + x0, i11 = y1 * WW + x1;

  size_t bhoff = (size_t)b * PP * CC + h * HDIM;
  const float4* q4  = (const float4*)(QT + bhoff + (size_t)p * CC);
  const float4* k00 = (const float4*)(KT + bhoff + (size_t)i00 * CC);
  const float4* k01 = (const float4*)(KT + bhoff + (size_t)i01 * CC);
  const float4* k10 = (const float4*)(KT + bhoff + (size_t)i10 * CC);
  const float4* k11 = (const float4*)(KT + bhoff + (size_t)i11 * CC);

  float score = 0.f;
#pragma unroll 4
  for (int d4 = 0; d4 < HDIM / 4; ++d4) {
    float4 a  = q4[d4];
    float4 v0 = k00[d4], v1 = k01[d4], v2 = k10[d4], v3 = k11[d4];
    score += a.x * (w00 * v0.x + w01 * v1.x + w10 * v2.x + w11 * v3.x);
    score += a.y * (w00 * v0.y + w01 * v1.y + w10 * v2.y + w11 * v3.y);
    score += a.z * (w00 * v0.z + w01 * v1.z + w10 * v2.z + w11 * v3.z);
    score += a.w * (w00 * v0.w + w01 * v1.w + w10 * v2.w + w11 * v3.w);
  }
  score *= SCALE;

  float mx = wred_max(score);
  float e = __expf(score - mx);
  float sum = wred_sum(e);
  float wgt = e / sum;

  const float4* v00 = (const float4*)(VT + bhoff + (size_t)i00 * CC);
  const float4* v01 = (const float4*)(VT + bhoff + (size_t)i01 * CC);
  const float4* v10 = (const float4*)(VT + bhoff + (size_t)i10 * CC);
  const float4* v11 = (const float4*)(VT + bhoff + (size_t)i11 * CC);
  float4* Ob = (float4*)(OT + bhoff + (size_t)p * CC);
#pragma unroll 4
  for (int d4 = 0; d4 < HDIM / 4; ++d4) {
    float4 a0 = v00[d4], a1 = v01[d4], a2 = v10[d4], a3 = v11[d4];
    float4 r;
    r.x = wgt * (w00 * a0.x + w01 * a1.x + w10 * a2.x + w11 * a3.x);
    r.y = wgt * (w00 * a0.y + w01 * a1.y + w10 * a2.y + w11 * a3.y);
    r.z = wgt * (w00 * a0.z + w01 * a1.z + w10 * a2.z + w11 * a3.z);
    r.w = wgt * (w00 * a0.w + w01 * a1.w + w10 * a2.w + w11 * a3.w);
    r = wred_sum4(r);
    if (lane == 0) Ob[d4] = r;
  }
}

// ---------------------------------------------------------------------------
// GroupNorm over (b, head) on pixel-major data, emit bf16 (pixel-major)
// ---------------------------------------------------------------------------
__global__ void k_gn(const float* __restrict__ OT, const float* __restrict__ gnw,
                     const float* __restrict__ gnb, bf16_t* __restrict__ GT) {
  __shared__ float rs_[256], rq_[256];
  int b = blockIdx.x >> 2, h = blockIdx.x & 3;
  size_t base = (size_t)b * PP * CC + h * HDIM;
  const int N = HDIM * PP;
  float s = 0.f, q2 = 0.f;
  for (int i = threadIdx.x; i < N; i += 256) {
    size_t addr = base + (size_t)(i >> 6) * CC + (i & 63);
    float v = OT[addr];
    s += v; q2 += v * v;
  }
  rs_[threadIdx.x] = s; rq_[threadIdx.x] = q2;
  __syncthreads();
  for (int o = 128; o > 0; o >>= 1) {
    if (threadIdx.x < o) {
      rs_[threadIdx.x] += rs_[threadIdx.x + o];
      rq_[threadIdx.x] += rq_[threadIdx.x + o];
    }
    __syncthreads();
  }
  float mu = rs_[0] * (1.f / N);
  float var = rq_[0] * (1.f / N) - mu * mu;
  float rstd = rsqrtf(var + GN_EPS);
  for (int i = threadIdx.x; i < N; i += 256) {
    int c = h * HDIM + (i & 63);
    size_t addr = base + (size_t)(i >> 6) * CC + (i & 63);
    float v = (OT[addr] - mu) * rstd * gnw[c] + gnb[c];
    GT[addr] = (bf16_t)v;
  }
}

// ---------------------------------------------------------------------------
// launch
// ---------------------------------------------------------------------------
extern "C" void kernel_launch(void* const* d_in, const int* in_sizes, int n_in,
                              void* d_out, int out_size, void* d_ws, size_t ws_size,
                              hipStream_t stream) {
  (void)in_sizes; (void)n_in; (void)out_size; (void)ws_size;
  const float* xf   = (const float*)d_in[0];
  const float* Wq   = (const float*)d_in[1];
  const float* Wk   = (const float*)d_in[2];
  const float* Wv   = (const float*)d_in[3];
  const float* Wo   = (const float*)d_in[4];
  const float* bo   = (const float*)d_in[5];
  const float* dirW = (const float*)d_in[6];
  const float* dirb = (const float*)d_in[7];
  const float* gnw  = (const float*)d_in[8];
  const float* gnb  = (const float*)d_in[9];
  float* out = (float*)d_out;

  char* ws = (char*)d_ws;
  size_t off = 0;
  auto walloc = [&](size_t bytes) -> void* {
    void* p = (void*)(ws + off);
    off += (bytes + 255) & ~(size_t)255;
    return p;
  };

  const size_t nBCP = (size_t)BB * CC * PP;
  bf16_t* xT   = (bf16_t*)walloc(nBCP * sizeof(bf16_t));   // [b,p,c] bf16
  bf16_t* wqb  = (bf16_t*)walloc((size_t)CC * CC * sizeof(bf16_t));
  bf16_t* wkb  = (bf16_t*)walloc((size_t)CC * CC * sizeof(bf16_t));
  bf16_t* wvb  = (bf16_t*)walloc((size_t)CC * CC * sizeof(bf16_t));
  bf16_t* wob  = (bf16_t*)walloc((size_t)CC * CC * sizeof(bf16_t));
  float*  QT   = (float*)walloc(nBCP * sizeof(float));     // [b,p,c]
  float*  KT   = (float*)walloc(nBCP * sizeof(float));
  float*  VT   = (float*)walloc(nBCP * sizeof(float));
  float*  OT   = (float*)walloc(nBCP * sizeof(float));
  float*  avg  = (float*)walloc((size_t)BB * CC * sizeof(float));
  float*  dirs = (float*)walloc((size_t)BB * NH * NS * 2 * sizeof(float));
  bf16_t* GT   = (bf16_t*)walloc(nBCP * sizeof(bf16_t));

  // staging: transpose x to pixel-major bf16; convert weights
  k_transpose_bf16<<<dim3(PP / 32, CC / 32, BB), dim3(32, 8), 0, stream>>>(xf, xT);
  k_f32_to_bf16<<<256, 256, 0, stream>>>(Wq, wqb, CC * CC);
  k_f32_to_bf16<<<256, 256, 0, stream>>>(Wk, wkb, CC * CC);
  k_f32_to_bf16<<<256, 256, 0, stream>>>(Wv, wvb, CC * CC);
  k_f32_to_bf16<<<256, 256, 0, stream>>>(Wo, wob, CC * CC);

  // direction prediction
  k_avg<<<BB * CC, 256, 0, stream>>>(xf, avg);
  k_dirs<<<1, 64, 0, stream>>>(avg, dirW, dirb, dirs);

  // Q/K/V projections (WMMA, pixel-major outputs)
  k_gemm_qkv<<<256, 256, 0, stream>>>(wqb, xT, QT);
  k_gemm_qkv<<<256, 256, 0, stream>>>(wkb, xT, KT);
  k_gemm_qkv<<<256, 256, 0, stream>>>(wvb, xT, VT);

  // deformable strip attention (vectorized gathers)
  k_attn<<<4096, 256, 0, stream>>>(QT, KT, VT, dirs, OT);

  // GroupNorm -> bf16 pixel-major (B operand layout for final GEMM)
  k_gn<<<8, 256, 0, stream>>>(OT, gnw, gnb, GT);

  // output projection + bias + residual (WMMA, channel-major store)
  k_gemm_out<<<256, 256, 0, stream>>>(wob, GT, bo, xf, out);
}
